// PiCANet_L_29042568855924
// MI455X (gfx1250) — matrix operand order repaired and checked by reading, not compiled, not run on hardware
//
#include <hip/hip_runtime.h>
#include <hip/hip_bf16.h>

// ---------------------------------------------------------------------------
// PiCANet-L attention block for MI455X (gfx1250, wave32, WMMA)
//   conv1 (256->128, 7x7 dil2 pad6)  : implicit-GEMM, v_wmma_f32_16x16x32_bf16
//                                      double-buffered LDS, 8 WMMA/wave/iter
//   conv2 (128->49, 1x1) + softmax   : VALU, thread-per-pixel
//   49-tap dilated weighted gather   : VALU
// ---------------------------------------------------------------------------

#define BATCH 4
#define CIN   256
#define Hdim  64
#define Wdim  64
#define OC1   128
#define OC2   49
#define PADR  6
#define HP    76          // 64 + 2*6
#define WP    76
#define KTOT  (49 * 256)  // 12544 = implicit-GEMM K
#define NCHUNK (KTOT / 32)
#define NPIX  (BATCH * Hdim * Wdim)      // 16384

#define AROWH 40                  // padded halves per LDS row (80B)
#define ATILE (64 * AROWH)        // halves per A buffer
#define BTILE (128 * AROWH)       // halves per B buffer

typedef __attribute__((ext_vector_type(16))) __bf16 v16bf;
typedef __attribute__((ext_vector_type(8)))  float  v8f;

union FragBF {
    uint4 u4[2];
    v16bf v;
};

// ---------------------------------------------------------------------------
// Prep kernel 0: x [4][256][64][64] f32  ->  xp [4][76][76][256] bf16, padded.
// ---------------------------------------------------------------------------
__global__ __launch_bounds__(256) void pad_nhwc_bf16(
    const float* __restrict__ x, __hip_bfloat16* __restrict__ xp) {
    const int total = BATCH * HP * WP * CIN;
    int idx = blockIdx.x * 256 + threadIdx.x;
    if (idx >= total) return;
    const int c  = idx & 255;
    int t        = idx >> 8;
    const int wp = t % WP; t /= WP;
    const int hp = t % HP;
    const int b  = t / HP;
    const int h = hp - PADR;
    const int w = wp - PADR;
    float v = 0.0f;
    if (h >= 0 && h < Hdim && w >= 0 && w < Wdim)
        v = x[(((size_t)b * CIN + c) << 12) + (h << 6) + w];
    xp[idx] = __float2bfloat16(v);
}

// ---------------------------------------------------------------------------
// Prep kernel 1: w1 [128][256][7][7] f32 -> w1b [oc][uv*256 + ic] bf16
// ---------------------------------------------------------------------------
__global__ __launch_bounds__(256) void wprep_bf16(
    const float* __restrict__ w1, __hip_bfloat16* __restrict__ w1b) {
    const int total = OC1 * 49 * CIN;
    int idx = blockIdx.x * 256 + threadIdx.x;
    if (idx >= total) return;
    const int ic = idx & 255;
    int t        = idx >> 8;
    const int uv = t % 49;
    const int oc = t / 49;
    w1b[idx] = __float2bfloat16(w1[((size_t)oc * CIN + ic) * 49 + uv]);
}

// ---------------------------------------------------------------------------
// conv1 as implicit GEMM:  k1[m][oc] = b1[oc] + sum_k A[m][k] * B[oc][k]
//   m = b*4096 + h*64 + w  (one block = one (b,h) row = 64 pixels)
//   k = uv*256 + ic        (A[m][k] = xp[b][h+2u][w+2v][ic], contiguous in ic)
// Block: 128 threads = 4 waves. Block tile 64M x 128N.
//   wave: m_pair = wave&1  -> M subtiles {2*m_pair, 2*m_pair+1} (32 rows)
//         n_half = wave>>1 -> 64 cols = 4 N subtiles
//   => 2x4 = 8 accumulators, 8 WMMAs per wave per K-step.
// Double-buffered LDS, one barrier per K-step; global loads for step kc+1
// are issued before the WMMA chain of step kc to hide latency.
// ---------------------------------------------------------------------------
__global__ __launch_bounds__(128) void conv1_wmma(
    const __hip_bfloat16* __restrict__ xp,   // [4][76][76][256] bf16
    const __hip_bfloat16* __restrict__ w1b,  // [128][12544] bf16
    const float* __restrict__ b1,            // [128]
    float* __restrict__ k1) {                // [16384][128] f32
    const int bid  = blockIdx.x;             // 0..255 -> (b,h)
    const int b    = bid >> 6;
    const int h    = bid & 63;
    const int tid  = threadIdx.x;
    const int wave = tid >> 5;
    const int lane = tid & 31;

    __shared__ __align__(16) unsigned short At[2 * ATILE];   // 2 x 5.0 KB
    __shared__ __align__(16) unsigned short Bt[2 * BTILE];   // 2 x 10.0 KB

    const int m_pair = wave & 1;
    const int n_half = wave >> 1;

    v8f acc[2][4] = {};

    // Global->LDS staging (128 threads):
    //   A tile 64 rows x 64B: thread -> row tid>>1, 32B at (tid&1)*16 halves
    //   B tile 128 rows x 64B: thread -> full row tid (4 x b128)
    const int a_row  = tid >> 1;
    const int a_colh = (tid & 1) * 16;
    const int b_row  = tid;

    const unsigned short* xpu = reinterpret_cast<const unsigned short*>(xp);
    const unsigned short* w1u = reinterpret_cast<const unsigned short*>(w1b);

    // Fragment read offsets (ISA 7.12.2 layouts):
    //   A: lanes 0-15 -> K {0-7,16-23}; lanes 16-31 -> K {8-15,24-31}
    //   B: lanes 0-15 -> K {0-7,16-23}; lanes 16-31 -> K {8-15,24-31}
    const int khA   = (lane >> 4) * 8;
    const int khB   = (lane >> 4) * 8;
    const int arow0 = (m_pair * 2) * 16 + (lane & 15);
    const int arow1 = arow0 + 16;

    // ---- preload chunk 0 into buffer 0 ----
    {
        const unsigned short* ap = xpu +
            ((((size_t)b * HP + h) * WP + (a_row)) * CIN + a_colh);   // uv=0,u=v=0,ic0=0
        const uint4 av0 = *reinterpret_cast<const uint4*>(ap);
        const uint4 av1 = *reinterpret_cast<const uint4*>(ap + 8);
        const unsigned short* bp = w1u + (size_t)b_row * KTOT;
        const uint4 bv0 = *reinterpret_cast<const uint4*>(bp);
        const uint4 bv1 = *reinterpret_cast<const uint4*>(bp + 8);
        const uint4 bv2 = *reinterpret_cast<const uint4*>(bp + 16);
        const uint4 bv3 = *reinterpret_cast<const uint4*>(bp + 24);
        *reinterpret_cast<uint4*>(&At[a_row * AROWH + a_colh])     = av0;
        *reinterpret_cast<uint4*>(&At[a_row * AROWH + a_colh + 8]) = av1;
        unsigned short* bd = &Bt[b_row * AROWH];
        *reinterpret_cast<uint4*>(bd)      = bv0;
        *reinterpret_cast<uint4*>(bd + 8)  = bv1;
        *reinterpret_cast<uint4*>(bd + 16) = bv2;
        *reinterpret_cast<uint4*>(bd + 24) = bv3;
    }
    __syncthreads();

    for (int kc = 0; kc < NCHUNK; ++kc) {
        const int cur = kc & 1;
        const int nxt = cur ^ 1;
        const bool more = (kc + 1) < NCHUNK;

        // Issue global loads for chunk kc+1 (in flight during the WMMA chain).
        uint4 av0, av1, bv0, bv1, bv2, bv3;
        if (more) {
            const int k0  = (kc + 1) * 32;
            const int uv  = k0 >> 8;          // chunk never crosses a (u,v) tap
            const int ic0 = k0 & 255;
            const int u   = uv / 7;
            const int v   = uv - u * 7;
            const unsigned short* ap = xpu +
                ((((size_t)b * HP + (h + 2 * u)) * WP + (a_row + 2 * v)) * CIN
                 + ic0 + a_colh);
            av0 = *reinterpret_cast<const uint4*>(ap);
            av1 = *reinterpret_cast<const uint4*>(ap + 8);
            const unsigned short* bp = w1u + (size_t)b_row * KTOT + k0;
            bv0 = *reinterpret_cast<const uint4*>(bp);
            bv1 = *reinterpret_cast<const uint4*>(bp + 8);
            bv2 = *reinterpret_cast<const uint4*>(bp + 16);
            bv3 = *reinterpret_cast<const uint4*>(bp + 24);
        }

        // ---- compute from buffer `cur` ----
        const unsigned short* Ac = At + cur * ATILE;
        const unsigned short* Bc = Bt + cur * BTILE;

        FragBF a0, a1;
        a0.u4[0] = *reinterpret_cast<const uint4*>(&Ac[arow0 * AROWH + khA]);
        a0.u4[1] = *reinterpret_cast<const uint4*>(&Ac[arow0 * AROWH + khA + 16]);
        a1.u4[0] = *reinterpret_cast<const uint4*>(&Ac[arow1 * AROWH + khA]);
        a1.u4[1] = *reinterpret_cast<const uint4*>(&Ac[arow1 * AROWH + khA + 16]);

#pragma unroll
        for (int nt = 0; nt < 4; ++nt) {
            FragBF bb;
            const int brow = n_half * 64 + nt * 16 + (lane & 15);
            bb.u4[0] = *reinterpret_cast<const uint4*>(&Bc[brow * AROWH + khB]);
            bb.u4[1] = *reinterpret_cast<const uint4*>(&Bc[brow * AROWH + khB + 16]);
            acc[0][nt] = __builtin_amdgcn_wmma_f32_16x16x32_bf16(
                false, a0.v, false, bb.v, (short)0, acc[0][nt], false, false);
            acc[1][nt] = __builtin_amdgcn_wmma_f32_16x16x32_bf16(
                false, a1.v, false, bb.v, (short)0, acc[1][nt], false, false);
        }

        // ---- stage chunk kc+1 into buffer `nxt` ----
        if (more) {
            unsigned short* An = At + nxt * ATILE;
            unsigned short* Bn = Bt + nxt * BTILE;
            *reinterpret_cast<uint4*>(&An[a_row * AROWH + a_colh])     = av0;
            *reinterpret_cast<uint4*>(&An[a_row * AROWH + a_colh + 8]) = av1;
            unsigned short* bd = &Bn[b_row * AROWH];
            *reinterpret_cast<uint4*>(bd)      = bv0;
            *reinterpret_cast<uint4*>(bd + 8)  = bv1;
            *reinterpret_cast<uint4*>(bd + 16) = bv2;
            *reinterpret_cast<uint4*>(bd + 24) = bv3;
        }
        __syncthreads();   // single barrier: covers store->read (nxt) and
                           // read->overwrite (cur) across the ping-pong
    }

    // Epilogue: fuse +b1[oc], store pixel-major k1[m][oc].
    // C/D layout: VGPR r -> M = r + 8*(lane>=16), N = lane%16.
#pragma unroll
    for (int s = 0; s < 2; ++s) {
        const int m_base = bid * 64 + (m_pair * 2 + s) * 16 + ((lane >> 4) << 3);
#pragma unroll
        for (int nt = 0; nt < 4; ++nt) {
            const int oc = n_half * 64 + nt * 16 + (lane & 15);
            const float bias = b1[oc];
#pragma unroll
            for (int r = 0; r < 8; ++r) {
                k1[(size_t)(m_base + r) * OC1 + oc] = acc[s][nt][r] + bias;
            }
        }
    }
}

// ---------------------------------------------------------------------------
// conv2 (1x1, 128->49) + bias + softmax over the 49 channels, per pixel.
// ---------------------------------------------------------------------------
__global__ __launch_bounds__(256) void conv2_softmax(
    const float* __restrict__ k1,   // [16384][128]
    const float* __restrict__ w2,   // [49][128]
    const float* __restrict__ b2,   // [49]
    float* __restrict__ attn) {     // [16384][49]
    const int m = blockIdx.x * 256 + threadIdx.x;
    if (m >= NPIX) return;

    float acc[OC2];
#pragma unroll
    for (int j = 0; j < OC2; ++j) acc[j] = b2[j];

    const float* row = k1 + (size_t)m * OC1;
    for (int ic = 0; ic < OC1; ++ic) {
        const float xv = row[ic];
#pragma unroll
        for (int j = 0; j < OC2; ++j)
            acc[j] = fmaf(xv, w2[j * OC1 + ic], acc[j]);
    }

    float mx = acc[0];
#pragma unroll
    for (int j = 1; j < OC2; ++j) mx = fmaxf(mx, acc[j]);
    float s = 0.0f;
#pragma unroll
    for (int j = 0; j < OC2; ++j) {
        acc[j] = __expf(acc[j] - mx);
        s += acc[j];
    }
    const float inv = 1.0f / s;
    float* arow = attn + (size_t)m * OC2;
#pragma unroll
    for (int j = 0; j < OC2; ++j) arow[j] = acc[j] * inv;
}

// ---------------------------------------------------------------------------
// out[b,c,h,w] = sum_{u,v} attn[m][u*7+v] * x[b,c,h+2u-6,w+2v-6]  (0 if OOB)
// ---------------------------------------------------------------------------
__global__ __launch_bounds__(256) void gather_out(
    const float* __restrict__ x,     // [4][256][64][64]
    const float* __restrict__ attn,  // [16384][49]
    float* __restrict__ out) {       // [4][256][64][64]
    const int total = BATCH * CIN * Hdim * Wdim;
    int idx = blockIdx.x * 256 + threadIdx.x;
    if (idx >= total) return;
    const int w = idx & 63;
    const int h = (idx >> 6) & 63;
    const int c = (idx >> 12) & 255;
    const int b = idx >> 20;
    const int m = ((b << 6) + h) * 64 + w;   // b*4096 + h*64 + w
    const float* arow = attn + (size_t)m * OC2;
    const float* xc   = x + (((size_t)b * CIN + c) << 12);

    float s = 0.0f;
#pragma unroll
    for (int u = 0; u < 7; ++u) {
        const int hh = h + 2 * u - PADR;
        if (hh < 0 || hh >= Hdim) continue;
#pragma unroll
        for (int v = 0; v < 7; ++v) {
            const int ww = w + 2 * v - PADR;
            if (ww < 0 || ww >= Wdim) continue;
            s = fmaf(arow[u * 7 + v], xc[(hh << 6) + ww], s);
        }
    }
    out[idx] = s;
}

// ---------------------------------------------------------------------------
// Launch
// ---------------------------------------------------------------------------
extern "C" void kernel_launch(void* const* d_in, const int* in_sizes, int n_in,
                              void* d_out, int out_size, void* d_ws, size_t ws_size,
                              hipStream_t stream) {
    (void)in_sizes; (void)n_in; (void)out_size; (void)ws_size;
    const float* x  = (const float*)d_in[0];
    const float* w1 = (const float*)d_in[1];
    const float* b1 = (const float*)d_in[2];
    const float* w2 = (const float*)d_in[3];
    const float* b2 = (const float*)d_in[4];

    // Workspace layout (bytes), 256B-aligned offsets:
    //   xp   bf16 [4][76][76][256]  @ 0          (11,829,248)
    //   w1b  bf16 [128][12544]      @ 11,829,248 ( 3,211,264)
    //   k1   f32  [16384][128]      @ 15,040,512 ( 8,388,608)
    //   attn f32  [16384][49]       @ 23,429,120 ( 3,211,264)  total ~26.6 MB
    char* ws = (char*)d_ws;
    __hip_bfloat16* xp  = (__hip_bfloat16*)(ws);
    __hip_bfloat16* w1b = (__hip_bfloat16*)(ws + 11829248);
    float* k1   = (float*)(ws + 15040512);
    float* attn = (float*)(ws + 23429120);
    float* out  = (float*)d_out;

    pad_nhwc_bf16<<<(BATCH * HP * WP * CIN + 255) / 256, 256, 0, stream>>>(x, xp);
    wprep_bf16<<<(OC1 * 49 * CIN + 255) / 256, 256, 0, stream>>>(w1, w1b);
    conv1_wmma<<<BATCH * Hdim, 128, 0, stream>>>(xp, w1b, b1, k1);
    conv2_softmax<<<NPIX / 256, 256, 0, stream>>>(k1, w2, b2, attn);
    gather_out<<<(BATCH * CIN * Hdim * Wdim) / 256, 256, 0, stream>>>(x, attn, out);
}
